// MultiheadAttention_55989193670756
// MI455X (gfx1250) — matrix operand order repaired and checked by reading, not compiled, not run on hardware
//
#include <hip/hip_runtime.h>
#include <hip/hip_bf16.h>

// ---------------- types ----------------
typedef __bf16 bf16;
typedef __attribute__((ext_vector_type(16))) __bf16 v16bf;
typedef __attribute__((ext_vector_type(8)))  __bf16 v8bf;
typedef __attribute__((ext_vector_type(8)))  float  v8f;

#define WMMA_BF16(a, b, c) \
  __builtin_amdgcn_wmma_f32_16x16x32_bf16(false, (a), false, (b), (short)0, (c), false, false)

// ---------------- problem constants ----------------
static constexpr int BB  = 2;
static constexpr int SS  = 4096;
static constexpr int DD  = 512;
static constexpr int HH  = 8;
static constexpr int HD  = 64;
static constexpr int NQ  = 3 * DD;          // 1536
static constexpr float NEGV = -9.0e15f;
static constexpr float SCALE = 0.125f;      // 1/sqrt(64)

// ---------------- workspace layout (bytes) ----------------
static constexpr size_t OFF_XB    = 0;                                   // [B*S][D] bf16
static constexpr size_t OFF_WQKVT = OFF_XB    + (size_t)BB*SS*DD*2;      // [3D][D]  bf16
static constexpr size_t OFF_WOT   = OFF_WQKVT + (size_t)NQ*DD*2;         // [D][D]   bf16
static constexpr size_t OFF_Q     = OFF_WOT   + (size_t)DD*DD*2;         // [B*H][S][HD] bf16
static constexpr size_t OFF_K     = OFF_Q     + (size_t)BB*HH*SS*HD*2;
static constexpr size_t OFF_VT    = OFF_K     + (size_t)BB*HH*SS*HD*2;   // [B*H][HD][S] bf16
static constexpr size_t OFF_VALS  = OFF_VT    + (size_t)BB*HH*SS*HD*2;   // [B*S][D] bf16

// ---------------- WMMA fragment loaders (wave32 layouts) ----------------
// A 16x32 bf16, row-major, leading dim lda.
// lane = half*16 + M ; a[0..7] = A[M][half*8 + j] ; a[8..15] = A[M][16 + half*8 + j]
__device__ __forceinline__ v16bf load_a_bf16(const bf16* A, int lda, int lane) {
  const int half = lane >> 4, m = lane & 15;
  const bf16* row = A + (size_t)m * lda;
  v8bf lo = *(const v8bf*)(row + half * 8);
  v8bf hi = *(const v8bf*)(row + 16 + half * 8);
  v16bf a;
#pragma unroll
  for (int j = 0; j < 8; ++j) { a[j] = lo[j]; a[j + 8] = hi[j]; }
  return a;
}

// Same A fragment but source is fp32 (attention probabilities), converted to bf16.
__device__ __forceinline__ v16bf load_a_f32(const float* A, int lda, int lane) {
  const int half = lane >> 4, m = lane & 15;
  const float* row = A + (size_t)m * lda;
  v8f lo = *(const v8f*)(row + half * 8);
  v8f hi = *(const v8f*)(row + 16 + half * 8);
  v16bf a;
#pragma unroll
  for (int j = 0; j < 8; ++j) { a[j] = (bf16)lo[j]; a[j + 8] = (bf16)hi[j]; }
  return a;
}

// B 32x16 bf16 from an N-major buffer BT[n][k] (so B[k][n] = BT[n][k]).
// lane = half*16 + N ; b[j] = BT[N][half*8 + j] ; b[8 + j] = BT[N][16 + half*8 + j]
__device__ __forceinline__ v16bf load_bt_bf16(const bf16* BT, int ldb, int lane) {
  const int half = lane >> 4, n = lane & 15;
  const bf16* row = BT + (size_t)n * ldb + half * 8;
  v8bf lo = *(const v8bf*)(row);
  v8bf hi = *(const v8bf*)(row + 16);
  v16bf b;
#pragma unroll
  for (int j = 0; j < 8; ++j) { b[j] = lo[j]; b[j + 8] = hi[j]; }
  return b;
}

// ---------------- conversion / transpose kernels ----------------
__global__ void k_cvt_bf16(const float* __restrict__ src, bf16* __restrict__ dst, int n) {
  int i = blockIdx.x * blockDim.x + threadIdx.x;
  if (i < n) dst[i] = (bf16)src[i];
}

// dst[n*K + k] = (bf16)src[k*N + n]   (src is K x N row-major)
__global__ void k_transpose_bf16(const float* __restrict__ src, bf16* __restrict__ dst,
                                 int K, int N) {
  int i = blockIdx.x * blockDim.x + threadIdx.x;
  if (i >= K * N) return;
  int k = i / N, n = i % N;
  dst[(size_t)n * K + k] = (bf16)src[i];
}

// ---------------- QKV projection: qkv = x @ w_qkv + b_qkv, scattered per-head ----------------
// M = B*S (8192), N = 3D (1536), K = D (512).  8 waves/block, one 16x16 tile per wave.
__global__ void k_qkv_gemm(const bf16* __restrict__ xb, const bf16* __restrict__ wqkvT,
                           const float* __restrict__ bias,
                           bf16* __restrict__ q, bf16* __restrict__ kbuf,
                           bf16* __restrict__ vT) {
  const int lane = threadIdx.x & 31;
  const int wave = threadIdx.x >> 5;
  const int n_tile = blockIdx.x * 8 + wave;   // 96 n-tiles
  const int m_tile = blockIdx.y;              // 512 m-tiles

  const bf16* A  = xb    + (size_t)m_tile * 16 * DD;
  const bf16* BT = wqkvT + (size_t)n_tile * 16 * DD;

  v8f acc = {};
  for (int kk = 0; kk < DD; kk += 32) {
    v16bf a = load_a_bf16(A + kk, DD, lane);
    v16bf b = load_bt_bf16(BT + kk, DD, lane);
    acc = WMMA_BF16(a, b, acc);
  }

  const int half = lane >> 4, nl = lane & 15;
  const int n = n_tile * 16 + nl;
  const float bn = bias[n];
  const int h = n / (3 * HD);
  const int r = n % (3 * HD);
#pragma unroll
  for (int ri = 0; ri < 8; ++ri) {
    const int m = m_tile * 16 + half * 8 + ri;
    const int bidx = m / SS, s = m % SS;
    const bf16 val = (bf16)(acc[ri] + bn);
    const size_t bh = (size_t)bidx * HH + h;
    if (r < HD)            q   [(bh * SS + s) * HD + r]            = val;
    else if (r < 2 * HD)   kbuf[(bh * SS + s) * HD + (r - HD)]     = val;
    else                   vT  [(bh * HD + (r - 2 * HD)) * SS + s] = val;
  }
}

// ---------------- logits = (Q @ K^T) * scale, masked, fp32 -> d_out attn region ----------------
// Per (b,h): M = S, N = S, K = HD (two WMMA steps).
__global__ void k_logits(const bf16* __restrict__ q, const bf16* __restrict__ kbuf,
                         const int* __restrict__ mask, float* __restrict__ attn) {
  const int lane = threadIdx.x & 31;
  const int wave = threadIdx.x >> 5;
  const int bh = blockIdx.z;
  const int bidx = bh / HH;
  const int m_tile = blockIdx.y;              // 256
  const int n_tile = blockIdx.x * 8 + wave;   // 256

  const bf16* Q  = q    + (size_t)bh * SS * HD + (size_t)m_tile * 16 * HD;
  const bf16* KT = kbuf + (size_t)bh * SS * HD + (size_t)n_tile * 16 * HD; // K row-major == B^T

  v8f acc = {};
#pragma unroll
  for (int kk = 0; kk < HD; kk += 32) {
    v16bf a = load_a_bf16(Q + kk, HD, lane);
    v16bf b = load_bt_bf16(KT + kk, HD, lane);
    acc = WMMA_BF16(a, b, acc);
  }

  const int half = lane >> 4, nl = lane & 15;
  const int n = n_tile * 16 + nl;
  const bool keep = (mask[bidx * SS + n] != 0);
  float* out = attn + ((size_t)bh * SS + m_tile * 16 + half * 8) * SS + n;
#pragma unroll
  for (int ri = 0; ri < 8; ++ri) {
    out[(size_t)ri * SS] = keep ? acc[ri] * SCALE : NEGV;
  }
}

// ---------------- in-place row softmax over last axis (row length S) ----------------
__global__ void k_softmax(float* __restrict__ attn) {
  const size_t row = blockIdx.x;
  float* p = attn + row * (size_t)SS;
  const int tid = threadIdx.x;
  __shared__ float red[256];

  float m = -INFINITY;
  for (int i = tid; i < SS; i += 256) m = fmaxf(m, p[i]);
  red[tid] = m; __syncthreads();
  for (int s = 128; s > 0; s >>= 1) { if (tid < s) red[tid] = fmaxf(red[tid], red[tid + s]); __syncthreads(); }
  m = red[0]; __syncthreads();

  float sum = 0.f;
  for (int i = tid; i < SS; i += 256) { float e = __expf(p[i] - m); p[i] = e; sum += e; }
  red[tid] = sum; __syncthreads();
  for (int s = 128; s > 0; s >>= 1) { if (tid < s) red[tid] += red[tid + s]; __syncthreads(); }
  const float inv = 1.0f / red[0];

  for (int i = tid; i < SS; i += 256) p[i] *= inv;
}

// ---------------- values = attn @ V ----------------
// Per (b,h): M = S, N = HD, K = S. Each wave owns a full 16x64 row-block
// (4 accumulators) so each attention row is read from HBM exactly once.
__global__ void k_values(const float* __restrict__ attn, const bf16* __restrict__ vT,
                         bf16* __restrict__ vals) {
  const int lane = threadIdx.x & 31;
  const int wave = threadIdx.x >> 5;
  const int bh = blockIdx.y;
  const int bidx = bh / HH, h = bh % HH;
  const int m_tile = blockIdx.x * 8 + wave;   // 256 m-tiles

  const float* A  = attn + ((size_t)bh * SS + m_tile * 16) * SS;
  const bf16*  VT = vT   + (size_t)bh * HD * SS;            // [HD][S] == B^T

  v8f acc0 = {}, acc1 = {}, acc2 = {}, acc3 = {};
  for (int kk = 0; kk < SS; kk += 32) {
    v16bf a = load_a_f32(A + kk, SS, lane);
    v16bf b0 = load_bt_bf16(VT + (size_t)0  * SS + kk, SS, lane);
    v16bf b1 = load_bt_bf16(VT + (size_t)16 * SS + kk, SS, lane);
    v16bf b2 = load_bt_bf16(VT + (size_t)32 * SS + kk, SS, lane);
    v16bf b3 = load_bt_bf16(VT + (size_t)48 * SS + kk, SS, lane);
    acc0 = WMMA_BF16(a, b0, acc0);
    acc1 = WMMA_BF16(a, b1, acc1);
    acc2 = WMMA_BF16(a, b2, acc2);
    acc3 = WMMA_BF16(a, b3, acc3);
  }

  const int half = lane >> 4, nl = lane & 15;
#pragma unroll
  for (int ri = 0; ri < 8; ++ri) {
    const int s = m_tile * 16 + half * 8 + ri;
    bf16* row = vals + ((size_t)bidx * SS + s) * DD + h * HD + nl;
    row[0]  = (bf16)acc0[ri];
    row[16] = (bf16)acc1[ri];
    row[32] = (bf16)acc2[ri];
    row[48] = (bf16)acc3[ri];
  }
}

// ---------------- output projection: o = vals @ w_o + b_o (fp32 out) ----------------
// M = B*S (8192), N = D (512), K = D (512).
__global__ void k_out_gemm(const bf16* __restrict__ vals, const bf16* __restrict__ woT,
                           const float* __restrict__ bias, float* __restrict__ o) {
  const int lane = threadIdx.x & 31;
  const int wave = threadIdx.x >> 5;
  const int n_tile = blockIdx.x * 8 + wave;   // 32 n-tiles
  const int m_tile = blockIdx.y;              // 512 m-tiles

  const bf16* A  = vals + (size_t)m_tile * 16 * DD;
  const bf16* BT = woT  + (size_t)n_tile * 16 * DD;

  v8f acc = {};
  for (int kk = 0; kk < DD; kk += 32) {
    v16bf a = load_a_bf16(A + kk, DD, lane);
    v16bf b = load_bt_bf16(BT + kk, DD, lane);
    acc = WMMA_BF16(a, b, acc);
  }

  const int half = lane >> 4, nl = lane & 15;
  const int n = n_tile * 16 + nl;
  const float bn = bias[n];
#pragma unroll
  for (int ri = 0; ri < 8; ++ri) {
    const int m = m_tile * 16 + half * 8 + ri;
    o[(size_t)m * DD + n] = acc[ri] + bn;
  }
}

// ---------------- host launcher ----------------
extern "C" void kernel_launch(void* const* d_in, const int* in_sizes, int n_in,
                              void* d_out, int out_size, void* d_ws, size_t ws_size,
                              hipStream_t stream) {
  const float* x     = (const float*)d_in[0];
  const int*   mask  = (const int*)  d_in[1];
  const float* w_qkv = (const float*)d_in[2];
  const float* b_qkv = (const float*)d_in[3];
  const float* w_o   = (const float*)d_in[4];
  const float* b_o   = (const float*)d_in[5];

  char* ws = (char*)d_ws;
  bf16* xb     = (bf16*)(ws + OFF_XB);
  bf16* wqkvT  = (bf16*)(ws + OFF_WQKVT);
  bf16* woT    = (bf16*)(ws + OFF_WOT);
  bf16* qbuf   = (bf16*)(ws + OFF_Q);
  bf16* kbuf   = (bf16*)(ws + OFF_K);
  bf16* vT     = (bf16*)(ws + OFF_VT);
  bf16* vals   = (bf16*)(ws + OFF_VALS);

  float* o    = (float*)d_out;                         // [B*S*D]
  float* attn = (float*)d_out + (size_t)BB * SS * DD;  // [B*H*S*S]

  // 1) precision conversion + weight transposes
  {
    int n = BB * SS * DD;
    k_cvt_bf16<<<(n + 255) / 256, 256, 0, stream>>>(x, xb, n);
    k_transpose_bf16<<<(DD * NQ + 255) / 256, 256, 0, stream>>>(w_qkv, wqkvT, DD, NQ);
    k_transpose_bf16<<<(DD * DD + 255) / 256, 256, 0, stream>>>(w_o, woT, DD, DD);
  }
  // 2) QKV projection
  k_qkv_gemm<<<dim3(NQ / 16 / 8, BB * SS / 16), 256, 0, stream>>>(
      xb, wqkvT, b_qkv, qbuf, kbuf, vT);
  // 3) masked, scaled logits -> d_out attention region
  k_logits<<<dim3(SS / 16 / 8, SS / 16, BB * HH), 256, 0, stream>>>(
      qbuf, kbuf, mask, attn);
  // 4) row softmax in place
  k_softmax<<<(size_t)BB * HH * SS, 256, 0, stream>>>(attn);
  // 5) attn @ V
  k_values<<<dim3(SS / 16 / 8, BB * HH), 256, 0, stream>>>(attn, vT, vals);
  // 6) output projection
  k_out_gemm<<<dim3(DD / 16 / 8, BB * SS / 16), 256, 0, stream>>>(vals, woT, b_o, o);
}
